// SeqAggregator_26963804685005
// MI455X (gfx1250) — compile-verified, matrix operand order checked by reading, not run on hardware
//
#include <hip/hip_runtime.h>
#include <hip/hip_bf16.h>

// CDNA5 / gfx1250: wave32, WMMA 16x16x32 bf16 with f32 accumulate.
typedef __bf16 v16bf __attribute__((ext_vector_type(16)));
typedef float  v8f   __attribute__((ext_vector_type(8)));

#define B_TOT   16384
#define T_SEQ   50
#define DN      128   // D_NEIGH
#define D_INV   256   // D_IN
#define D_OUTV  256   // D_OUT
#define H_DIM   128
#define G_DIM   512   // 4H

// LDS row strides padded to avoid bank conflicts on the per-lane A gathers
#define XLD   132
#define HLD   132
#define HLLD  136
#define SLD   260

__device__ __forceinline__ float sigm(float x) { return 1.0f / (1.0f + __expf(-x)); }

__device__ __forceinline__ float tanh_fast(float x) {
  float ax = fabsf(x);
  float e  = __expf(-2.0f * ax);
  float r  = (1.0f - e) / (1.0f + e);
  return copysignf(r, x);
}

__device__ __forceinline__ v8f wmma_bf16(v16bf a, v16bf b, v8f c) {
  // (neg_a, A, neg_b, B, c_mod, C, reuse_a, reuse_b)
  return __builtin_amdgcn_wmma_f32_16x16x32_bf16(false, a, false, b, (short)0, c,
                                                 false, false);
}

// K-position of packed pair v (0..7) inside a 16-bit 16x32 operand fragment:
// VGPR 0..3 -> K = 2v + 8*hi ; VGPR 4..7 -> K = 16 + 2(v-4) + 8*hi
__device__ __forceinline__ int kpat(int v, int hi) {
  return (v < 4) ? (v * 2 + hi * 8) : (16 + (v - 4) * 2 + hi * 8);
}

// A fragment (16x32 bf16) gathered from an f32 row-major LDS tile [16 x >=k0+32]
__device__ __forceinline__ v16bf a_frag_f32(const float* buf, int ld, int k0, int lane) {
  int row = lane & 15, hi = lane >> 4;
  v16bf a;
#pragma unroll
  for (int v = 0; v < 8; ++v) {
    const float* p = buf + row * ld + k0 + kpat(v, hi);
    a[2 * v]     = (__bf16)p[0];
    a[2 * v + 1] = (__bf16)p[1];
  }
  return a;
}

// A fragment gathered from a bf16 row-major LDS tile
__device__ __forceinline__ v16bf a_frag_bf16(const __bf16* buf, int ld, int k0, int lane) {
  int row = lane & 15, hi = lane >> 4;
  v16bf a;
#pragma unroll
  for (int v = 0; v < 8; ++v) {
    const __bf16* p = buf + row * ld + k0 + kpat(v, hi);
    a[2 * v]     = p[0];
    a[2 * v + 1] = p[1];
  }
  return a;
}

// B fragment (32x16 bf16, K-striped like A, N = lane%16) from a global
// row-major f32 weight matrix W[K][ldn], tile at (k0, col0).
__device__ __forceinline__ v16bf b_frag_global(const float* W, int ldn, int k0,
                                               int col0, int lane) {
  int col = col0 + (lane & 15), hi = lane >> 4;
  v16bf b;
#pragma unroll
  for (int v = 0; v < 8; ++v) {
    int k = k0 + kpat(v, hi);
    b[2 * v]     = (__bf16)W[(size_t)k * ldn + col];
    b[2 * v + 1] = (__bf16)W[(size_t)(k + 1) * ldn + col];
  }
  return b;
}

__global__ __launch_bounds__(256) void SeqAggregator_26963804685005_kernel(
    const float* __restrict__ self_vecs,  // [B, 256]
    const float* __restrict__ neigh,      // [B, T, 128]
    const float* __restrict__ Wx,         // [128, 512]
    const float* __restrict__ Wh,         // [128, 512]
    const float* __restrict__ b_lstm,     // [512]
    const float* __restrict__ Wn,         // [128, 256]
    const float* __restrict__ Ws,         // [256, 256]
    float* __restrict__ out) {            // [B, 256]
  __shared__ float  xbuf[16 * XLD];       // x_t tile (f32)
  __shared__ float  hbuf[16 * HLD];       // h state (f32)
  __shared__ float  cbuf[16 * H_DIM];     // c state (f32)
  __shared__ float  gates[16 * G_DIM];    // gate pre-activations / self tile
  __shared__ __bf16 hlast[16 * HLLD];     // h at last valid timestep (bf16)
  __shared__ int    rownz[16];

  const int tid  = threadIdx.x;
  const int wave = tid >> 5;
  const int lane = tid & 31;
  const int hi   = lane >> 4;
  const int nsub = lane & 15;
  const int b0   = blockIdx.x * 16;

  // ---- Weights resident in VGPRs: each wave owns 4 N-tiles (64 gate cols) ----
  v16bf wxB[4][4], whB[4][4];
  float bl[4];
#pragma unroll
  for (int nt = 0; nt < 4; ++nt) {
    int col0 = (wave * 4 + nt) * 16;
    bl[nt] = b_lstm[col0 + nsub];
#pragma unroll
    for (int kc = 0; kc < 4; ++kc) {
      wxB[nt][kc] = b_frag_global(Wx, G_DIM, kc * 32, col0, lane);
      whB[nt][kc] = b_frag_global(Wh, G_DIM, kc * 32, col0, lane);
    }
  }

  // ---- zero LSTM state ----
  for (int e = tid; e < 16 * HLD; e += 256) hbuf[e] = 0.0f;
  for (int e = tid; e < 16 * H_DIM; e += 256) cbuf[e] = 0.0f;
  __syncthreads();

  // ---- 50 LSTM steps ----
  for (int t = 0; t < T_SEQ; ++t) {
    const float* xg = neigh + (size_t)b0 * T_SEQ * DN + (size_t)t * DN;
    for (int e = tid; e < 16 * DN; e += 256) {
      int m = e >> 7, n = e & 127;
      xbuf[m * XLD + n] = xg[(size_t)m * T_SEQ * DN + n];
      if (t + 1 < T_SEQ)  // prefetch next timestep's rows -> global_prefetch_b8
        __builtin_prefetch(xg + (size_t)m * T_SEQ * DN + DN + n, 0, 1);
    }
    __syncthreads();

    // per-row "nonzero" flag (prefix mask => last nonzero row == length-1)
    if (tid < 16) {
      float mx = 0.0f;
      for (int n = 0; n < DN; ++n) mx = fmaxf(mx, fabsf(xbuf[tid * XLD + n]));
      rownz[tid] = (mx > 0.0f) ? 1 : 0;
    }

    // gather A fragments once (shared across this wave's 4 N-tiles)
    v16bf ax[4], ah[4];
#pragma unroll
    for (int kc = 0; kc < 4; ++kc) {
      ax[kc] = a_frag_f32(xbuf, XLD, kc * 32, lane);
      ah[kc] = a_frag_f32(hbuf, HLD, kc * 32, lane);
    }

#pragma unroll
    for (int nt = 0; nt < 4; ++nt) {
      v8f acc = {};
#pragma unroll
      for (int kc = 0; kc < 4; ++kc) acc = wmma_bf16(ax[kc], wxB[nt][kc], acc);
#pragma unroll
      for (int kc = 0; kc < 4; ++kc) acc = wmma_bf16(ah[kc], whB[nt][kc], acc);
      int col0 = (wave * 4 + nt) * 16;
#pragma unroll
      for (int r = 0; r < 8; ++r)
        gates[(r + hi * 8) * G_DIM + col0 + nsub] = acc[r] + bl[nt];
    }
    __syncthreads();

    // elementwise LSTM cell (TF BasicLSTMCell: i, j, f, o ; forget bias 1.0)
    for (int e = tid; e < 16 * H_DIM; e += 256) {
      int m = e >> 7, n = e & 127;
      const float* g = gates + m * G_DIM;
      float iv = g[n], jv = g[H_DIM + n], fv = g[2 * H_DIM + n], ov = g[3 * H_DIM + n];
      float c = cbuf[e];
      c = c * sigm(fv + 1.0f) + sigm(iv) * tanh_fast(jv);
      float h = tanh_fast(c) * sigm(ov);
      cbuf[e] = c;
      hbuf[m * HLD + n] = h;
      if (t == 0 || rownz[m]) hlast[m * HLLD + n] = (__bf16)h;
    }
    __syncthreads();
  }

  // ---- out = relu(self @ Ws + h_last @ Wn) ----
  for (int e = tid; e < 16 * D_INV; e += 256) {
    int m = e >> 8, n = e & 255;
    gates[m * SLD + n] = self_vecs[(size_t)(b0 + m) * D_INV + n];
  }
  __syncthreads();

#pragma unroll
  for (int part = 0; part < 2; ++part) {
    int col0 = (wave + part * 8) * 16;
    v8f acc = {};
#pragma unroll
    for (int kc = 0; kc < 8; ++kc)  // self @ Ws, K = 256
      acc = wmma_bf16(a_frag_f32(gates, SLD, kc * 32, lane),
                      b_frag_global(Ws, D_OUTV, kc * 32, col0, lane), acc);
#pragma unroll
    for (int kc = 0; kc < 4; ++kc)  // h_last @ Wn, K = 128
      acc = wmma_bf16(a_frag_bf16(hlast, HLLD, kc * 32, lane),
                      b_frag_global(Wn, D_OUTV, kc * 32, col0, lane), acc);
#pragma unroll
    for (int r = 0; r < 8; ++r) {
      int m = r + hi * 8;
      float v = acc[r];
      out[(size_t)(b0 + m) * D_OUTV + col0 + nsub] = v > 0.0f ? v : 0.0f;
    }
  }
}

extern "C" void kernel_launch(void* const* d_in, const int* in_sizes, int n_in,
                              void* d_out, int out_size, void* d_ws, size_t ws_size,
                              hipStream_t stream) {
  (void)in_sizes; (void)n_in; (void)out_size; (void)d_ws; (void)ws_size;
  const float* self_vecs = (const float*)d_in[0];
  const float* neigh     = (const float*)d_in[1];
  // d_in[2] = temperature (unused by the reference computation)
  const float* Wx        = (const float*)d_in[3];
  const float* Wh        = (const float*)d_in[4];
  const float* bl        = (const float*)d_in[5];
  const float* Wn        = (const float*)d_in[6];  // neigh_weights
  const float* Ws        = (const float*)d_in[7];  // self_weights
  float* out = (float*)d_out;

  dim3 grid(B_TOT / 16);
  dim3 block(256);
  hipLaunchKernelGGL(SeqAggregator_26963804685005_kernel, grid, block, 0, stream,
                     self_vecs, neigh, Wx, Wh, bl, Wn, Ws, out);
}
